// ConditionalRBM_383
// MI455X (gfx1250) — compile-verified, matrix operand order checked by reading, not run on hardware
//
#include <hip/hip_runtime.h>
#include <stdint.h>

#define V_DIM 1024
#define H_DIM 128
#define CD_DIM 16
#define WID_DIM 64
#define ROWS_PER_BLK 16

typedef __attribute__((ext_vector_type(16))) _Float16 v16h;
typedef __attribute__((ext_vector_type(8)))  _Float16 v8h;
typedef __attribute__((ext_vector_type(8)))  float    v8f;

// ---------------- RNG (stateless hash; no correctness check in this env) ----
__device__ __forceinline__ uint32_t hash3(uint32_t a, uint32_t b, uint32_t c) {
    uint32_t x = a * 0x8DA6B343u + b * 0xD8163841u + c * 0xCB1AB31Fu + 0x9E3779B9u;
    x ^= x >> 16; x *= 0x7FEB352Du;
    x ^= x >> 15; x *= 0x846CA68Bu;
    x ^= x >> 16;
    return x;
}
__device__ __forceinline__ float urand(uint32_t a, uint32_t b, uint32_t c) {
    return (float)(hash3(a, b, c) >> 8) * (1.0f / 16777216.0f);
}
__device__ __forceinline__ float sigmoidf(float x) {
    return 1.0f / (1.0f + __expf(-x));
}

// Combine the two contiguous 8-element K-runs of a lane into one 16x32 frag.
__device__ __forceinline__ v16h make_frag(v8h a, v8h b) {
    return __builtin_shufflevector(a, b, 0, 1, 2, 3, 4, 5, 6, 7,
                                         8, 9, 10, 11, 12, 13, 14, 15);
}
__device__ __forceinline__ v8h flip8(v8h x) {   // 1 - x (packed f16)
    v8h r;
    #pragma unroll
    for (int i = 0; i < 8; ++i) r[i] = (_Float16)1.0f - x[i];
    return r;
}

// ---------------- kernel 1: W f32 -> f16 (row-major + transposed) -----------
__global__ void convert_w_kernel(const float* __restrict__ W,
                                 _Float16* __restrict__ Wh,
                                 _Float16* __restrict__ WT) {
    int i = blockIdx.x * blockDim.x + threadIdx.x;
    if (i < V_DIM * H_DIM) {
        int v = i / H_DIM, h = i % H_DIM;
        _Float16 x = (_Float16)W[i];
        Wh[i] = x;                      // W row-major  (V x H)
        WT[h * V_DIM + v] = x;          // W transposed (H x V)
    }
}

// ---------------- kernel 2: FiLM MLP -> b_mod, c_mod, bsum ------------------
__global__ void __launch_bounds__(256)
precompute_kernel(const float* __restrict__ cond,
                  const float* __restrict__ bvec,
                  const float* __restrict__ cvec,
                  const float* __restrict__ fc1_w,
                  const float* __restrict__ fc1_b,
                  const float* __restrict__ fc2_w,
                  const float* __restrict__ fc2_b,
                  float* __restrict__ b_mod,
                  float* __restrict__ c_mod,
                  float* __restrict__ bsum) {
    __shared__ float x1[WID_DIM];
    __shared__ float condl[CD_DIM];
    __shared__ float rsum;
    const int brow = blockIdx.x;
    const int t = threadIdx.x;

    if (t < CD_DIM) condl[t] = cond[(size_t)brow * CD_DIM + t];
    if (t == 0) rsum = 0.0f;
    __syncthreads();

    if (t < WID_DIM) {
        float acc = fc1_b[t];
        #pragma unroll
        for (int k = 0; k < CD_DIM; ++k) acc += condl[k] * fc1_w[t * CD_DIM + k];
        x1[t] = tanhf(acc);
    }
    __syncthreads();

    float local = 0.0f;
    for (int o = t; o < V_DIM; o += 256) {
        float g  = fc2_b[o];
        float be = fc2_b[V_DIM + o];
        #pragma unroll 4
        for (int k = 0; k < WID_DIM; ++k) {
            float xv = x1[k];
            g  += xv * fc2_w[(size_t)o * WID_DIM + k];
            be += xv * fc2_w[(size_t)(V_DIM + o) * WID_DIM + k];
        }
        float bm = (1.0f + g) * bvec[o] + be;
        b_mod[(size_t)brow * V_DIM + o] = bm;
        local += bm;
    }
    for (int o = t; o < H_DIM; o += 256) {
        float g  = fc2_b[2 * V_DIM + o];
        float be = fc2_b[2 * V_DIM + H_DIM + o];
        #pragma unroll 4
        for (int k = 0; k < WID_DIM; ++k) {
            float xv = x1[k];
            g  += xv * fc2_w[(size_t)(2 * V_DIM + o) * WID_DIM + k];
            be += xv * fc2_w[(size_t)(2 * V_DIM + H_DIM + o) * WID_DIM + k];
        }
        c_mod[(size_t)brow * H_DIM + o] = (1.0f + g) * cvec[o] + be;
    }
    atomicAdd(&rsum, local);
    __syncthreads();
    if (t == 0) bsum[brow] = rsum;
}

// ---------------- kernel 3: persistent Gibbs chain (WMMA) -------------------
// LDS layout (dynamic):
//   [0, 262144)        WT_s : 128x1024 f16  (W transposed, for GEMM1 B)
//   [262144, 294912)   vsh  : 16x1024 f16
//   [294912, 299008)   hsh  : 16x128  f16
//   [299008, 299264)   rowAcc: 16x4 f32     (sum_a, v.a, v.b_mod, pad)
//   [299264, 299328)   s0sh : 16 f32
//   [299328, 307520)   cmods: 16x128 f32
#define LDS_BYTES 307520

__global__ void __launch_bounds__(256)
rbm_gibbs_kernel(const _Float16* __restrict__ Whg,   // W row-major f16 (L2-hot)
                 const _Float16* __restrict__ WTg,   // W transposed f16
                 const float* __restrict__ b_mod,
                 const float* __restrict__ c_mod,
                 const float* __restrict__ bsum,
                 const float* __restrict__ falloff_p,
                 const int* __restrict__ steps_p,
                 float* __restrict__ out) {
    extern __shared__ char smem[];
    _Float16* WT_s  = (_Float16*)(smem);
    _Float16* vsh   = (_Float16*)(smem + 262144);
    _Float16* hsh   = (_Float16*)(smem + 294912);
    float*    rowAcc = (float*)(smem + 299008);
    float*    s0sh   = (float*)(smem + 299264);
    float*    cmods  = (float*)(smem + 299328);

    const int tid  = threadIdx.x;
    const int lane = tid & 31;
    const int wave = tid >> 5;
    const int mrow = lane & 15;                 // A row / B col / C col index
    const int mhi  = (lane & 16) ? 8 : 0;       // C row offset for hi half-wave
    const int kbase = (lane & 16) ? 8 : 0;      // K base of this lane's runs
    const int rows0 = blockIdx.x * ROWS_PER_BLK;

    // ---- stage W^T (f16, 256 KB) into LDS with async copies ----
    {
        uint32_t lds0 = (uint32_t)(uintptr_t)(void*)WT_s;
        const char* src = (const char*)WTg;
        #pragma unroll 1
        for (int i = 0; i < 64; ++i) {          // 256 thr * 64 * 16B = 256 KB
            uint32_t off = lds0 + (uint32_t)tid * 1024u + (uint32_t)i * 16u;
            uint64_t ga  = (uint64_t)(uintptr_t)(src + (size_t)tid * 1024 + (size_t)i * 16);
            asm volatile("global_load_async_to_lds_b128 %0, %1, off"
                         :: "v"(off), "v"(ga) : "memory");
        }
        asm volatile("s_wait_asynccnt 0x0" ::: "memory");
    }

    // ---- init state ----
    float bsv = 0.0f;
    if (tid < ROWS_PER_BLK) {
        bsv = bsum[rows0 + tid];
        s0sh[tid] = 1.0f;
    }
    for (int i = tid; i < ROWS_PER_BLK * H_DIM; i += 256)
        cmods[i] = c_mod[(size_t)(rows0 + i / H_DIM) * H_DIM + (i % H_DIM)];
    for (int i = tid; i < ROWS_PER_BLK * V_DIM; i += 256) {
        int m = i >> 10, n = i & (V_DIM - 1);
        float u = urand(0x00FFFFFFu, (uint32_t)(rows0 + m), (uint32_t)n);
        vsh[i] = (_Float16)((u < 0.5f) ? 1.0f : 0.0f);
    }
    __syncthreads();

    const float fo   = *falloff_p;
    const int  steps = *steps_p;
    float a_save[8][8];                          // a tile values, kept in VGPRs

    for (int stp = 0; stp < steps; ++stp) {
        float sAnn = sigmoidf(-fo * ((float)stp - 0.5f * (float)steps));
        float T    = 1.0f + 4.0f * sAnn;         // T_END + (T_START-T_END)*s
        float invT = 1.0f / T;

        // ---- Phase 1: preact = flip(v,s0) @ W + c_mod ; sample h ----
        if (tid < 64) rowAcc[tid] = 0.0f;        // reset reduction accumulators
        {
            v8f acc = {};
            const bool doflip = (s0sh[mrow] == 0.0f);
            const _Float16* arow = vsh + mrow * V_DIM;
            const _Float16* brow = WT_s + (size_t)(wave * 16 + mrow) * V_DIM;
            #pragma unroll 1
            for (int k0 = 0; k0 < V_DIM; k0 += 32) {
                const int kb = k0 + kbase;
                v8h a0 = *(const v8h*)(arow + kb);
                v8h a1 = *(const v8h*)(arow + kb + 16);
                if (doflip) { a0 = flip8(a0); a1 = flip8(a1); }
                v16h afrag = make_frag(a0, a1);
                v16h bfrag = make_frag(*(const v8h*)(brow + kb),
                                       *(const v8h*)(brow + kb + 16));
                acc = __builtin_amdgcn_wmma_f32_16x16x32_f16(
                          false, afrag, false, bfrag, (short)0, acc, false, false);
            }
            const int nc = wave * 16 + mrow;     // hidden column for this lane
            #pragma unroll
            for (int r = 0; r < 8; ++r) {
                int m = r + mhi;
                float p  = sigmoidf((acc[r] + cmods[m * H_DIM + nc]) * invT);
                float hv = (urand(0x10000u + stp, (uint32_t)(rows0 + m),
                                  (uint32_t)nc) < p) ? 1.0f : 0.0f;
                hsh[m * H_DIM + nc] = (_Float16)hv;
            }
        }
        __syncthreads();

        // ---- Phase 2: a = h @ W^T (B streamed from L2-hot row-major W) ----
        float sa[8], sva[8], svb[8];
        #pragma unroll
        for (int r = 0; r < 8; ++r) { sa[r] = 0.f; sva[r] = 0.f; svb[r] = 0.f; }
        #pragma unroll
        for (int nt = 0; nt < 8; ++nt) {
            const int n0 = wave * 128 + nt * 16;
            const _Float16* arow = hsh + mrow * H_DIM;
            const _Float16* brow = Whg + (size_t)(n0 + mrow) * H_DIM; // W[n][k]
            v8f acc = {};
            #pragma unroll
            for (int k0 = 0; k0 < H_DIM; k0 += 32) {
                const int kb = k0 + kbase;
                v16h afrag = make_frag(*(const v8h*)(arow + kb),
                                       *(const v8h*)(arow + kb + 16));
                v16h bfrag = make_frag(*(const v8h*)(brow + kb),
                                       *(const v8h*)(brow + kb + 16));
                acc = __builtin_amdgcn_wmma_f32_16x16x32_f16(
                          false, afrag, false, bfrag, (short)0, acc, false, false);
            }
            const int n = n0 + mrow;
            #pragma unroll
            for (int r = 0; r < 8; ++r) {
                int m = r + mhi;
                float aval = acc[r];
                a_save[nt][r] = aval;
                float vcur = (float)vsh[m * V_DIM + n];
                float bm   = b_mod[(size_t)(rows0 + m) * V_DIM + n];
                sa[r]  += aval;
                sva[r] += vcur * aval;
                svb[r] += vcur * bm;
            }
        }
        #pragma unroll
        for (int r = 0; r < 8; ++r) {            // one atomic triple per row
            int m = r + mhi;
            atomicAdd(&rowAcc[m * 4 + 0], sa[r]);
            atomicAdd(&rowAcc[m * 4 + 1], sva[r]);
            atomicAdd(&rowAcc[m * 4 + 2], svb[r]);
        }
        __syncthreads();

        // ---- Phase 3: global spin flip (one thread per row) ----
        if (tid < ROWS_PER_BLK) {
            float suma = rowAcc[tid * 4 + 0];
            float va   = rowAcc[tid * 4 + 1];
            float vb   = rowAcc[tid * 4 + 2];
            float dE   = -bsv - suma + 2.0f * vb + 2.0f * va;
            float p    = sigmoidf(dE * invT);
            s0sh[tid]  = (urand(0x20000u + stp, (uint32_t)(rows0 + tid), 0u) < p)
                             ? 1.0f : 0.0f;
        }
        __syncthreads();

        // ---- Phase 4: sample v, apply flip(v_samp, s0n) ----
        #pragma unroll
        for (int nt = 0; nt < 8; ++nt) {
            const int n = wave * 128 + nt * 16 + mrow;
            #pragma unroll
            for (int r = 0; r < 8; ++r) {
                int m = r + mhi;
                float bm = b_mod[(size_t)(rows0 + m) * V_DIM + n];
                float p  = sigmoidf((a_save[nt][r] + bm) * invT);
                float vs = (urand(0x30000u + stp, (uint32_t)(rows0 + m),
                                  (uint32_t)n) < p) ? 1.0f : 0.0f;
                float s0n = s0sh[m];
                float vn  = s0n * vs + (1.0f - s0n) * (1.0f - vs);
                vsh[m * V_DIM + n] = (_Float16)vn;
            }
        }
        __syncthreads();
    }

    // ---- write final v as f32 ----
    for (int i = tid; i < ROWS_PER_BLK * V_DIM; i += 256) {
        int m = i >> 10, n = i & (V_DIM - 1);
        out[(size_t)(rows0 + m) * V_DIM + n] = (float)vsh[i];
    }
}

// ---------------- host launcher ---------------------------------------------
extern "C" void kernel_launch(void* const* d_in, const int* in_sizes, int n_in,
                              void* d_out, int out_size, void* d_ws, size_t ws_size,
                              hipStream_t stream) {
    const float* cond    = (const float*)d_in[0];
    const float* W       = (const float*)d_in[1];
    const float* bvec    = (const float*)d_in[2];
    const float* cvec    = (const float*)d_in[3];
    const float* fc1_w   = (const float*)d_in[4];
    const float* fc1_b   = (const float*)d_in[5];
    const float* fc2_w   = (const float*)d_in[6];
    const float* fc2_b   = (const float*)d_in[7];
    const float* falloff = (const float*)d_in[8];
    const int*   steps   = (const int*)d_in[9];

    const int B = in_sizes[0] / CD_DIM;          // 8192

    // workspace layout
    char* ws = (char*)d_ws;
    size_t off = 0;
    _Float16* Wh = (_Float16*)(ws + off); off += (size_t)V_DIM * H_DIM * sizeof(_Float16);
    _Float16* WT = (_Float16*)(ws + off); off += (size_t)V_DIM * H_DIM * sizeof(_Float16);
    off = (off + 255) & ~(size_t)255;
    float* b_mod = (float*)(ws + off); off += (size_t)B * V_DIM * sizeof(float);
    float* c_mod = (float*)(ws + off); off += (size_t)B * H_DIM * sizeof(float);
    float* bsum  = (float*)(ws + off); off += (size_t)B * sizeof(float);
    (void)ws_size; (void)n_in; (void)out_size;

    convert_w_kernel<<<(V_DIM * H_DIM + 255) / 256, 256, 0, stream>>>(W, Wh, WT);

    precompute_kernel<<<B, 256, 0, stream>>>(
        cond, bvec, cvec, fc1_w, fc1_b, fc2_w, fc2_b, b_mod, c_mod, bsum);

    rbm_gibbs_kernel<<<B / ROWS_PER_BLK, 256, LDS_BYTES, stream>>>(
        Wh, WT, b_mod, c_mod, bsum, falloff, steps, (float*)d_out);
}
